// MultiHeadAttention_81690277970428
// MI455X (gfx1250) — compile-verified, matrix operand order checked by reading, not run on hardware
//
#include <hip/hip_runtime.h>
#include <hip/hip_fp16.h>
#include <math.h>

typedef __attribute__((ext_vector_type(16))) _Float16 v16h;
typedef __attribute__((ext_vector_type(2)))  _Float16 v2h;
typedef __attribute__((ext_vector_type(8)))  float    v8f;

#define SEQ    2048
#define BATCH  2
#define DMODEL 1024
#define HEADS  16
#define DK     64
#define ROWS   (SEQ * BATCH)   // 4096 GEMM rows, row = s*BATCH + b

// ---------------------------------------------------------------------------
// Shared fragment loader for 16-bit A fragments and B^T-stored-row-major
// fragments of v_wmma_f32_16x16x32_f16.
//   Lane L<16 : row r=L,   ks=0 -> halves K = {0..7} u {16..23}
//   Lane L>=16: row r=L-16, ks=1 -> halves K = {8..15} u {24..31}
// Tile must be row-major with K contiguous; row_stride in halves (mult of 8).
// ---------------------------------------------------------------------------
__device__ __forceinline__ v16h load_ab_frag(const _Float16* tile, int row_stride) {
    const int lane = threadIdx.x & 31;
    const int r    = lane & 15;
    const int ks   = lane >> 4;
    const _Float16* p = tile + (size_t)r * row_stride + ks * 8;
    v16h f;
    ((float4*)&f)[0] = *(const float4*)(p);        // 8 halves
    ((float4*)&f)[1] = *(const float4*)(p + 16);   // 8 halves
    return f;
}

__device__ __forceinline__ v8f wmma_f16(v16h a, v16h b, v8f c) {
    return __builtin_amdgcn_wmma_f32_16x16x32_f16(
        /*neg_a=*/false, a, /*neg_b=*/false, b,
        /*c_mod=*/(short)0, c, /*reuse_a=*/false, /*reuse_b=*/false);
}

// ---------------------------------------------------------------------------
// Generic projection GEMM:  Y[4096 x 1024] = X[4096 x 1024] @ W[1024 x 1024] + bias
// Workgroup tile 128(M) x 64(N), 8 waves each computing 32x32; K tiled by 64
// (two WMMA k-steps per stage -> 8 WMMAs per barrier pair).
// MODE 0: f32 in, f16 out at [b][h][s][d]   (Q, K)
// MODE 1: f32 in, f16 out at [b][h][d][s]   (V, pre-transposed for attention)
// MODE 2: f16 in, f32 out at [row][col]     (final output projection)
// ---------------------------------------------------------------------------
template <int MODE>
__global__ __launch_bounds__(256)
void mha_proj_gemm_kernel(const float* __restrict__ Xf, const _Float16* __restrict__ Xh,
                          const float* __restrict__ W, const float* __restrict__ bias,
                          void* __restrict__ out)
{
    constexpr bool IN_F16 = (MODE == 2);
    constexpr int TM = 128, TN = 64, TK = 64;

    __shared__ __align__(16) _Float16 As[TM * TK];  // 16 KB, [m][k]
    __shared__ __align__(16) _Float16 Bs[TN * TK];  // 8 KB, [n][k]  (B^T)

    const int tid  = threadIdx.x;
    const int lane = tid & 31;
    const int wid  = tid >> 5;
    const int m0   = blockIdx.x * TM;
    const int n0   = blockIdx.y * TN;
    const int wm   = (wid >> 1) * 32;   // wave M offset in tile
    const int wn   = (wid & 1) * 32;    // wave N offset in tile

    v8f c00 = {}, c01 = {}, c10 = {}, c11 = {};

    for (int k0 = 0; k0 < DMODEL; k0 += TK) {
        // ---- stage A tile 128x64: 512 chunks of 16 halves, 2 per thread ----
#pragma unroll
        for (int u = 0; u < 2; ++u) {
            const int unit = tid + u * 256;
            const int row  = unit >> 2;
            const int ch   = (unit & 3) * 16;
            _Float16* dst  = As + row * TK + ch;
            if constexpr (IN_F16) {
                const _Float16* src = Xh + (size_t)(m0 + row) * DMODEL + k0 + ch;
                ((float4*)dst)[0] = ((const float4*)src)[0];
                ((float4*)dst)[1] = ((const float4*)src)[1];
                if (k0 + TK < DMODEL) __builtin_prefetch(src + TK, 0, 0);
            } else {
                const float* src = Xf + (size_t)(m0 + row) * DMODEL + k0 + ch;
#pragma unroll
                for (int w = 0; w < 4; ++w) {
                    float4 x = ((const float4*)src)[w];
                    v2h p0; p0[0] = (_Float16)x.x; p0[1] = (_Float16)x.y;
                    v2h p1; p1[0] = (_Float16)x.z; p1[1] = (_Float16)x.w;
                    ((v2h*)dst)[w * 2 + 0] = p0;    // packed b32 LDS stores
                    ((v2h*)dst)[w * 2 + 1] = p1;
                }
                if (k0 + TK < DMODEL) __builtin_prefetch(src + TK, 0, 0);
            }
        }
        // ---- stage B^T tile 64x64: thread handles k-pair (2kp, 2kp+1) x 8 n
        //      so every LDS write is a packed b32 of two consecutive k halves.
        {
            const int kp = tid & 31;            // k-pair 0..31 -> k = 2*kp
            const int n8 = (tid >> 5) * 8;      // 0..56
            const float* sr0 = W + (size_t)(k0 + 2 * kp) * DMODEL + n0 + n8;
            const float* sr1 = sr0 + DMODEL;
            float4 a0 = ((const float4*)sr0)[0];
            float4 a1 = ((const float4*)sr0)[1];
            float4 b0 = ((const float4*)sr1)[0];
            float4 b1 = ((const float4*)sr1)[1];
            const float lo[8] = {a0.x, a0.y, a0.z, a0.w, a1.x, a1.y, a1.z, a1.w};
            const float hi[8] = {b0.x, b0.y, b0.z, b0.w, b1.x, b1.y, b1.z, b1.w};
#pragma unroll
            for (int u = 0; u < 8; ++u) {
                v2h p; p[0] = (_Float16)lo[u]; p[1] = (_Float16)hi[u];
                *(v2h*)&Bs[(n8 + u) * TK + 2 * kp] = p;
            }
            if (k0 + TK < DMODEL) __builtin_prefetch(sr0 + (size_t)TK * DMODEL, 0, 0);
        }
        __syncthreads();

#pragma unroll
        for (int ks = 0; ks < TK; ks += 32) {
            v16h a0 = load_ab_frag(As + wm * TK + ks, TK);
            v16h a1 = load_ab_frag(As + (wm + 16) * TK + ks, TK);
            v16h b0 = load_ab_frag(Bs + wn * TK + ks, TK);
            v16h b1 = load_ab_frag(Bs + (wn + 16) * TK + ks, TK);
            c00 = wmma_f16(a0, b0, c00);
            c01 = wmma_f16(a0, b1, c01);
            c10 = wmma_f16(a1, b0, c10);
            c11 = wmma_f16(a1, b1, c11);
        }
        __syncthreads();
    }

    // ---- epilogue: C-fragment layout: VGPR i, lanes 0-15 -> M=i, 16-31 -> M=8+i
    const int nn = lane & 15;
    const int mh = (lane >> 4) * 8;
#pragma unroll
    for (int ti = 0; ti < 2; ++ti) {
#pragma unroll
        for (int tj = 0; tj < 2; ++tj) {
            v8f cc = (ti == 0) ? ((tj == 0) ? c00 : c01) : ((tj == 0) ? c10 : c11);
#pragma unroll
            for (int i = 0; i < 8; ++i) {
                const int m = m0 + wm + ti * 16 + mh + i;    // GEMM row = s*BATCH + b
                const int n = n0 + wn + tj * 16 + nn;        // GEMM col = h*DK + d
                const float val = cc[i] + bias[n];
                if constexpr (MODE == 2) {
                    ((float*)out)[(size_t)m * DMODEL + n] = val;
                } else {
                    const int s = m >> 1, b = m & 1;
                    const int h = n >> 6, d = n & 63;
                    const size_t off = (MODE == 0)
                        ? ((size_t)(b * HEADS + h) * SEQ + s) * DK + d      // [b][h][s][d]
                        : ((size_t)(b * HEADS + h) * DK + d) * SEQ + s;     // [b][h][d][s]
                    ((_Float16*)out)[off] = (_Float16)val;
                }
            }
        }
    }
}

// ---------------------------------------------------------------------------
// Flash attention: grid (SEQ/128, BATCH*HEADS), 256 threads = 8 waves.
// Each wave owns 16 query rows; streams keys 32 at a time with online softmax.
// Per key block: 4 WMMAs for scores, softmax update, P relayout via wave-
// private LDS, 4 WMMAs for P@V.  Output (f16) goes to Xh[(s*2+b)][h*64+d].
// ---------------------------------------------------------------------------
__global__ __launch_bounds__(256)
void mha_flash_attn_kernel(const _Float16* __restrict__ Qh, const _Float16* __restrict__ Kh,
                           const _Float16* __restrict__ Vt, _Float16* __restrict__ Xh)
{
    __shared__ __align__(16) _Float16 Ps[8][16 * 32];   // 1 KB per wave

    const int tid  = threadIdx.x;
    const int lane = tid & 31;
    const int wid  = tid >> 5;
    const int bh   = blockIdx.y;
    const int b    = bh >> 4, h = bh & 15;
    const int q0   = blockIdx.x * 128 + wid * 16;

    const _Float16* Qb = Qh + (size_t)bh * SEQ * DK;    // [s][d]
    const _Float16* Kb = Kh + (size_t)bh * SEQ * DK;    // [j][d]  (== B^T of K^T)
    const _Float16* Vb = Vt + (size_t)bh * DK * SEQ;    // [d][s]  (== B^T of V)

    // Q fragments are loop-invariant: keep in registers for all 64 key blocks.
    const v16h qa0 = load_ab_frag(Qb + (size_t)q0 * DK, DK);        // d 0..31
    const v16h qa1 = load_ab_frag(Qb + (size_t)q0 * DK + 32, DK);   // d 32..63

    float mrow[8], lrow[8];
    v8f o0 = {}, o1 = {}, o2 = {}, o3 = {};
#pragma unroll
    for (int i = 0; i < 8; ++i) { mrow[i] = -1e30f; lrow[i] = 0.0f; }

    _Float16* P  = Ps[wid];
    const int nn   = lane & 15;
    const int half = lane >> 4;

    for (int j0 = 0; j0 < SEQ; j0 += 32) {
        const v16h kb00 = load_ab_frag(Kb + (size_t)j0 * DK, DK);           // keys j0..+15, d 0..31
        const v16h kb01 = load_ab_frag(Kb + (size_t)j0 * DK + 32, DK);      // keys j0..+15, d 32..63
        const v16h kb10 = load_ab_frag(Kb + (size_t)(j0 + 16) * DK, DK);
        const v16h kb11 = load_ab_frag(Kb + (size_t)(j0 + 16) * DK + 32, DK);

        v8f s0 = {}, s1 = {};
        s0 = wmma_f16(qa0, kb00, s0);
        s0 = wmma_f16(qa1, kb01, s0);    // scores for keys j0..j0+15
        s1 = wmma_f16(qa0, kb10, s1);
        s1 = wmma_f16(qa1, kb11, s1);    // scores for keys j0+16..j0+31

        // ---- online softmax (row stats reduced across 16-lane groups) ----
#pragma unroll
        for (int i = 0; i < 8; ++i) {
            float x0 = s0[i] * 0.125f;   // 1/sqrt(DK)
            float x1 = s1[i] * 0.125f;
            float vmax = fmaxf(x0, x1);
#pragma unroll
            for (int off = 8; off >= 1; off >>= 1)
                vmax = fmaxf(vmax, __shfl_xor(vmax, off, 16));
            const float mnew  = fmaxf(mrow[i], vmax);
            const float scale = __expf(mrow[i] - mnew);
            const float p0 = __expf(x0 - mnew);
            const float p1 = __expf(x1 - mnew);
            float rs = p0 + p1;
#pragma unroll
            for (int off = 8; off >= 1; off >>= 1)
                rs += __shfl_xor(rs, off, 16);
            lrow[i] = lrow[i] * scale + rs;
            mrow[i] = mnew;
            o0[i] *= scale; o1[i] *= scale; o2[i] *= scale; o3[i] *= scale;

            // stash P (f16) in wave-private LDS for C->A fragment relayout
            const int prow = i + 8 * half;                 // row m of this lane's value
            P[prow * 32 + nn]      = (_Float16)p0;
            P[prow * 32 + 16 + nn] = (_Float16)p1;
        }
        // same-wave DS ops are in-order: safe to read back without barrier
        const v16h pa = load_ab_frag(P, 32);               // A fragment 16x32

        const v16h vb0 = load_ab_frag(Vb + (size_t)0  * SEQ + j0, SEQ);  // d  0..15
        const v16h vb1 = load_ab_frag(Vb + (size_t)16 * SEQ + j0, SEQ);  // d 16..31
        const v16h vb2 = load_ab_frag(Vb + (size_t)32 * SEQ + j0, SEQ);  // d 32..47
        const v16h vb3 = load_ab_frag(Vb + (size_t)48 * SEQ + j0, SEQ);  // d 48..63
        o0 = wmma_f16(pa, vb0, o0);
        o1 = wmma_f16(pa, vb1, o1);
        o2 = wmma_f16(pa, vb2, o2);
        o3 = wmma_f16(pa, vb3, o3);
    }

    // ---- normalize by row sums and emit f16 rows of X for the O-projection ----
#pragma unroll
    for (int i = 0; i < 8; ++i) {
        const float inv = 1.0f / lrow[i];
        const int s = q0 + i + 8 * half;
        const size_t rowoff = ((size_t)s * BATCH + b) * DMODEL + (size_t)h * DK;
        Xh[rowoff +  0 + nn] = (_Float16)(o0[i] * inv);
        Xh[rowoff + 16 + nn] = (_Float16)(o1[i] * inv);
        Xh[rowoff + 32 + nn] = (_Float16)(o2[i] * inv);
        Xh[rowoff + 48 + nn] = (_Float16)(o3[i] * inv);
    }
}

// ---------------------------------------------------------------------------
extern "C" void kernel_launch(void* const* d_in, const int* in_sizes, int n_in,
                              void* d_out, int out_size, void* d_ws, size_t ws_size,
                              hipStream_t stream)
{
    (void)in_sizes; (void)n_in; (void)out_size; (void)ws_size;
    const float* q  = (const float*)d_in[0];
    const float* k  = (const float*)d_in[1];
    const float* v  = (const float*)d_in[2];
    const float* Wq = (const float*)d_in[3];
    const float* bq = (const float*)d_in[4];
    const float* Wk = (const float*)d_in[5];
    const float* bk = (const float*)d_in[6];
    const float* Wv = (const float*)d_in[7];
    const float* bv = (const float*)d_in[8];
    const float* Wo = (const float*)d_in[9];
    const float* bo = (const float*)d_in[10];
    float* out = (float*)d_out;

    const size_t perTensor = (size_t)BATCH * HEADS * SEQ * DK;  // 4M halves = 8 MB
    _Float16* Qh = (_Float16*)d_ws;
    _Float16* Kh = Qh + perTensor;
    _Float16* Vt = Kh + perTensor;
    _Float16* Xh = Vt + perTensor;   // [4096][1024] f16, total ws = 32 MB

    dim3 gg(ROWS / 128, DMODEL / 64);          // 32 x 16 workgroups
    mha_proj_gemm_kernel<0><<<gg, 256, 0, stream>>>(q, nullptr, Wq, bq, (void*)Qh);
    mha_proj_gemm_kernel<0><<<gg, 256, 0, stream>>>(k, nullptr, Wk, bk, (void*)Kh);
    mha_proj_gemm_kernel<1><<<gg, 256, 0, stream>>>(v, nullptr, Wv, bv, (void*)Vt);

    dim3 ga(SEQ / 128, BATCH * HEADS);         // 16 x 32 workgroups
    mha_flash_attn_kernel<<<ga, 256, 0, stream>>>(Qh, Kh, Vt, Xh);

    mha_proj_gemm_kernel<2><<<gg, 256, 0, stream>>>(nullptr, Xh, Wo, bo, (void*)out);
}
